// DeformableTransformerEncoderLayer_80427557585591
// MI455X (gfx1250) — compile-verified, matrix operand order checked by reading, not run on hardware
//
#include <hip/hip_runtime.h>
#include <hip/hip_bf16.h>

// ---------------------------------------------------------------------------
// DeformableTransformerEncoderLayer for gfx1250 (MI455X, wave32, WMMA)
// GEMMs: f16 inputs, f32 accumulation via V_WMMA_F32_16X16X32_F16.
// Each wave computes a 16x64 output tile (4 WMMA accumulators) so one A
// fragment feeds 4 WMMAs and the dependent-WMMA hazard is hidden by
// interleaving independent accumulator chains. No LDS/barriers in the GEMM:
// the whole working set (~85 MB) is L2-resident on a 192 MB L2.
// ---------------------------------------------------------------------------

typedef _Float16 half_t;
typedef __attribute__((ext_vector_type(16))) _Float16 v16h;
typedef __attribute__((ext_vector_type(8)))  _Float16 v8h;
typedef __attribute__((ext_vector_type(8)))  float    v8f;

static constexpr int  kL   = 8500;   // 80*80 + 40*40 + 20*20 + 10*10
static constexpr int  kD   = 256;
static constexpr int  kNH  = 8;
static constexpr int  kDFF = 1024;
static constexpr int  kMTiles = (kL + 15) / 16;   // 532

// ---------------------------------------------------------------------------
// Pack f32 weights (K x N, row-major) into wave32 WMMA B-fragment order:
// per (n_tile, k_block): 32 lanes x 16 halves, contiguous.
// Lane half H (=lane>>4), element e:
//   k = kb*32 + (e<8 ? H*8+e : 16 + H*8 + (e-8)),  n = ntile*16 + (lane&15)
// ---------------------------------------------------------------------------
__global__ __launch_bounds__(256)
void pack_b_kernel(const float* __restrict__ W, half_t* __restrict__ out,
                   int K, int N)
{
    int idx = blockIdx.x * 256 + threadIdx.x;
    if (idx >= K * N) return;
    const int KB       = K >> 5;
    const int e        = idx & 15;
    const int lane     = (idx >> 4) & 31;
    const int kb       = (idx >> 9) % KB;
    const int ntile    = (idx >> 9) / KB;
    const int laneHalf = lane >> 4;
    const int k = kb * 32 + ((e < 8) ? (laneHalf * 8 + e)
                                     : (16 + laneHalf * 8 + (e - 8)));
    const int n = ntile * 16 + (lane & 15);
    out[idx] = (half_t)W[(size_t)k * N + n];
}

// ---------------------------------------------------------------------------
// q = src + pos (f16) and src copy (f16) for the value projection
// ---------------------------------------------------------------------------
__global__ __launch_bounds__(256)
void addcvt_kernel(const float* __restrict__ src, const float* __restrict__ pos,
                   half_t* __restrict__ qf16, half_t* __restrict__ sf16, int n)
{
    int i = blockIdx.x * 256 + threadIdx.x;
    if (i < n) {
        float s = src[i];
        qf16[i] = (half_t)(s + pos[i]);
        sf16[i] = (half_t)s;
    }
}

// ---------------------------------------------------------------------------
// WMMA GEMM: C[M,N] = A[M,K] (f16, row-major) * Bpacked + bias
// One wave -> 16 rows x 64 cols (4 accumulators). Waves flattened over
// (m_tile, n_group). EPI = 0: f32 store. EPI = 1: ReLU + f16 store.
// ---------------------------------------------------------------------------
template <int EPI>
__global__ __launch_bounds__(256)
void gemm_wmma_kernel(const half_t* __restrict__ A,
                      const half_t* __restrict__ Bp,
                      const float*  __restrict__ bias,
                      void* __restrict__ Cout,
                      int M, int N, int K)
{
    const int lane     = threadIdx.x & 31;
    const int laneHalf = lane >> 4;
    const int lmod     = lane & 15;
    const int NG       = N >> 6;                 // 64-col groups
    const int mtiles   = (M + 15) >> 4;
    const int wgid     = blockIdx.x * 8 + (threadIdx.x >> 5);
    if (wgid >= mtiles * NG) return;
    const int mtile  = wgid / NG;
    const int ngroup = wgid % NG;
    const int m0     = mtile * 16;
    const int KB     = K >> 5;

    // A fragment rows: lane&15 selects row, lane>>4 selects K half.
    const int  arow    = m0 + lmod;
    const bool arow_ok = (arow < M);
    const half_t* ap   = A + (size_t)arow * K + laneHalf * 8;
    const half_t* bp   = Bp + ((size_t)(ngroup * 4) * KB + 0) * 512 + (size_t)lane * 16;

    v8f acc[4] = {v8f{}, v8f{}, v8f{}, v8f{}};

    #pragma unroll 2
    for (int kb = 0; kb < KB; ++kb) {
        v8h lo = {}, hi = {};
        if (arow_ok) {
            lo = *(const v8h*)(ap + kb * 32);
            hi = *(const v8h*)(ap + kb * 32 + 16);
        }
        const v16h a = __builtin_shufflevector(lo, hi, 0, 1, 2, 3, 4, 5, 6, 7,
                                                       8, 9, 10, 11, 12, 13, 14, 15);
        #pragma unroll
        for (int t = 0; t < 4; ++t) {
            // packed B: (((ngroup*4 + t)*KB + kb)*32 + lane) * 16 halves
            const v16h b = *(const v16h*)(bp + ((size_t)t * KB + kb) * 512);
            acc[t] = __builtin_amdgcn_wmma_f32_16x16x32_f16(
                         false, a, false, b, (short)0, acc[t], false, false);
        }
    }

    #pragma unroll
    for (int t = 0; t < 4; ++t) {
        const int col = (ngroup * 4 + t) * 16 + lmod;
        const float bcol = bias[col];
        #pragma unroll
        for (int r = 0; r < 8; ++r) {
            const int row = m0 + r + 8 * laneHalf;
            if (row < M) {
                float v = acc[t][r] + bcol;
                if (EPI == 0) {
                    ((float*)Cout)[(size_t)row * N + col] = v;
                } else {
                    ((half_t*)Cout)[(size_t)row * N + col] =
                        (half_t)(v > 0.f ? v : 0.f);
                }
            }
        }
    }
}

// ---------------------------------------------------------------------------
// Softmax over 16 logits per (l, h); logits layout (L, NH*16) contiguous.
// ---------------------------------------------------------------------------
__global__ __launch_bounds__(256)
void softmax16_kernel(float* __restrict__ logits, int n)
{
    int i = blockIdx.x * 256 + threadIdx.x;
    if (i >= n) return;
    float* p = logits + (size_t)i * 16;
    float m = p[0];
    #pragma unroll
    for (int j = 1; j < 16; ++j) m = fmaxf(m, p[j]);
    float s = 0.f;
    #pragma unroll
    for (int j = 0; j < 16; ++j) { float e = __expf(p[j] - m); p[j] = e; s += e; }
    const float inv = 1.f / s;
    #pragma unroll
    for (int j = 0; j < 16; ++j) p[j] *= inv;
}

// ---------------------------------------------------------------------------
// Deformable attention core: one wave per (query, head); lane = channel
// (DH == 32 == wave32). Bilinear weights are wave-uniform scalars; the
// per-corner gathers are 32x4 B coalesced lines from `value`.
// ---------------------------------------------------------------------------
__global__ __launch_bounds__(256)
void deform_sample_kernel(const float* __restrict__ value,
                          const float* __restrict__ off,
                          const float* __restrict__ attn,
                          const float* __restrict__ refp,
                          half_t* __restrict__ out)
{
    const int wid = blockIdx.x * 8 + (threadIdx.x >> 5);
    if (wid >= kL * kNH) return;
    const int l = wid >> 3;
    const int h = wid & 7;
    const int c = threadIdx.x & 31;

    const int Wl[4] = {80, 40, 20, 10};
    const int Hl[4] = {80, 40, 20, 10};
    const int st[4] = {0, 6400, 8000, 8400};

    float acc = 0.f;
    #pragma unroll
    for (int lvl = 0; lvl < 4; ++lvl) {
        const float rx = refp[(l * 4 + lvl) * 2 + 0];
        const float ry = refp[(l * 4 + lvl) * 2 + 1];
        const float Wf = (float)Wl[lvl], Hf = (float)Hl[lvl];
        #pragma unroll
        for (int p = 0; p < 4; ++p) {
            const int ob = l * 256 + h * 32 + lvl * 8 + p * 2;
            const float x = rx * Wf + off[ob + 0] - 0.5f;
            const float y = ry * Hf + off[ob + 1] - 0.5f;
            const float x0 = floorf(x), y0 = floorf(y);
            const float aw = attn[l * 128 + h * 16 + lvl * 4 + p];
            float samp = 0.f;
            #pragma unroll
            for (int dy = 0; dy < 2; ++dy) {
                #pragma unroll
                for (int dx = 0; dx < 2; ++dx) {
                    const float xi = x0 + (float)dx, yi = y0 + (float)dy;
                    const float w = (1.f - fabsf(x - xi)) * (1.f - fabsf(y - yi));
                    const bool valid = (xi >= 0.f) && (xi < Wf) &&
                                       (yi >= 0.f) && (yi < Hf);
                    const int xc = (int)fminf(fmaxf(xi, 0.f), Wf - 1.f);
                    const int yc = (int)fminf(fmaxf(yi, 0.f), Hf - 1.f);
                    const int lin = st[lvl] + yc * Wl[lvl] + xc;
                    const float g = value[(size_t)lin * 256 + h * 32 + c];
                    samp += valid ? (w * g) : 0.f;
                }
            }
            acc += aw * samp;
        }
    }
    out[(size_t)l * 256 + h * 32 + c] = (half_t)acc;
}

// ---------------------------------------------------------------------------
// Fused residual + LayerNorm: out = LN(a + b) * g + be. One wave per row,
// wave32 shuffle reduction. WRITE_H also stores an f16 copy for the next GEMM.
// ---------------------------------------------------------------------------
template <int WRITE_H>
__global__ __launch_bounds__(256)
void ln_kernel(const float* __restrict__ a, const float* __restrict__ b,
               const float* __restrict__ g, const float* __restrict__ be,
               float* __restrict__ out, half_t* __restrict__ outh, int M)
{
    const int row = blockIdx.x * 8 + (threadIdx.x >> 5);
    if (row >= M) return;
    const int lane = threadIdx.x & 31;
    const float* pa = a + (size_t)row * 256;
    const float* pb = b + (size_t)row * 256;

    float vals[8];
    float s = 0.f, sq = 0.f;
    #pragma unroll
    for (int j = 0; j < 8; ++j) {
        const int col = lane + j * 32;
        const float v = pa[col] + pb[col];
        vals[j] = v; s += v; sq += v * v;
    }
    #pragma unroll
    for (int o = 16; o > 0; o >>= 1) {
        s  += __shfl_xor(s,  o, 32);
        sq += __shfl_xor(sq, o, 32);
    }
    const float m   = s * (1.f / 256.f);
    const float var = sq * (1.f / 256.f) - m * m;
    const float rs  = rsqrtf(var + 1e-5f);
    #pragma unroll
    for (int j = 0; j < 8; ++j) {
        const int col = lane + j * 32;
        const float o2 = (vals[j] - m) * rs * g[col] + be[col];
        out[(size_t)row * 256 + col] = o2;
        if (WRITE_H) outh[(size_t)row * 256 + col] = (half_t)o2;
    }
}

// ---------------------------------------------------------------------------
// Host-side orchestration
// ---------------------------------------------------------------------------
extern "C" void kernel_launch(void* const* d_in, const int* in_sizes, int n_in,
                              void* d_out, int out_size, void* d_ws, size_t ws_size,
                              hipStream_t stream)
{
    (void)in_sizes; (void)n_in; (void)out_size; (void)ws_size;

    const float* src  = (const float*)d_in[0];
    const float* pos  = (const float*)d_in[1];
    const float* refp = (const float*)d_in[2];
    // d_in[3] spatial_shapes, d_in[4] level_start_index, d_in[5] scale: compile-time
    const float* Wo   = (const float*)d_in[6];
    const float* bo   = (const float*)d_in[7];
    const float* Wa   = (const float*)d_in[8];
    const float* ba   = (const float*)d_in[9];
    const float* Wv   = (const float*)d_in[10];
    const float* bv   = (const float*)d_in[11];
    const float* Wout = (const float*)d_in[12];
    const float* bout = (const float*)d_in[13];
    const float* W1   = (const float*)d_in[14];
    const float* b1   = (const float*)d_in[15];
    const float* W2   = (const float*)d_in[16];
    const float* b2   = (const float*)d_in[17];
    const float* g1   = (const float*)d_in[18];
    const float* be1  = (const float*)d_in[19];
    const float* g2   = (const float*)d_in[20];
    const float* be2  = (const float*)d_in[21];
    float* out = (float*)d_out;

    // Workspace carve (256 B aligned)
    char* w = (char*)d_ws;
    auto alloc = [&](size_t bytes) -> void* {
        void* p = (void*)w;
        w += (bytes + 255) & ~(size_t)255;
        return p;
    };
    half_t* qf16   = (half_t*)alloc((size_t)kL * kD * 2);
    half_t* sf16   = (half_t*)alloc((size_t)kL * kD * 2);
    half_t* WoP    = (half_t*)alloc((size_t)256 * 256 * 2);
    half_t* WaP    = (half_t*)alloc((size_t)256 * 128 * 2);
    half_t* WvP    = (half_t*)alloc((size_t)256 * 256 * 2);
    half_t* WoutP  = (half_t*)alloc((size_t)256 * 256 * 2);
    half_t* W1P    = (half_t*)alloc((size_t)256 * 1024 * 2);
    half_t* W2P    = (half_t*)alloc((size_t)1024 * 256 * 2);
    float*  offb   = (float*) alloc((size_t)kL * 256 * 4);
    float*  attnb  = (float*) alloc((size_t)kL * 128 * 4);
    float*  valb   = (float*) alloc((size_t)kL * 256 * 4);
    half_t* aoutH  = (half_t*)alloc((size_t)kL * 256 * 2);
    float*  src2b  = (float*) alloc((size_t)kL * 256 * 4);
    float*  xb     = (float*) alloc((size_t)kL * 256 * 4);
    half_t* xf16   = (half_t*)alloc((size_t)kL * 256 * 2);
    half_t* hb     = (half_t*)alloc((size_t)kL * kDFF * 2);
    float*  ffnb   = (float*) alloc((size_t)kL * 256 * 4);

    // 1) pack all weights into WMMA B-fragment layout
    pack_b_kernel<<<(256 * 256)  / 256, 256, 0, stream>>>(Wo,   WoP,   256, 256);
    pack_b_kernel<<<(256 * 128)  / 256, 256, 0, stream>>>(Wa,   WaP,   256, 128);
    pack_b_kernel<<<(256 * 256)  / 256, 256, 0, stream>>>(Wv,   WvP,   256, 256);
    pack_b_kernel<<<(256 * 256)  / 256, 256, 0, stream>>>(Wout, WoutP, 256, 256);
    pack_b_kernel<<<(256 * 1024) / 256, 256, 0, stream>>>(W1,   W1P,   256, 1024);
    pack_b_kernel<<<(1024 * 256) / 256, 256, 0, stream>>>(W2,   W2P,   1024, 256);

    // 2) q = src + pos (f16); src copy (f16)
    {
        const int n = kL * kD;
        addcvt_kernel<<<(n + 255) / 256, 256, 0, stream>>>(src, pos, qf16, sf16, n);
    }

    // GEMM grids: waves = mtiles * (N/64), 8 waves per block
    auto blocks_for = [](int N) { return (kMTiles * (N >> 6) + 7) / 8; };

    // 3) projection GEMMs
    gemm_wmma_kernel<0><<<blocks_for(256),  256, 0, stream>>>(qf16, WoP, bo, offb,  kL, 256, 256);
    gemm_wmma_kernel<0><<<blocks_for(128),  256, 0, stream>>>(qf16, WaP, ba, attnb, kL, 128, 256);
    gemm_wmma_kernel<0><<<blocks_for(256),  256, 0, stream>>>(sf16, WvP, bv, valb,  kL, 256, 256);

    // 4) softmax over the 16 (level, point) logits per (query, head)
    softmax16_kernel<<<(kL * kNH + 255) / 256, 256, 0, stream>>>(attnb, kL * kNH);

    // 5) deformable bilinear sampling -> f16 attention output
    deform_sample_kernel<<<(kL * kNH + 7) / 8, 256, 0, stream>>>(valb, offb, attnb, refp, aoutH);

    // 6) output projection, residual + LN1 (f32 + f16 copies)
    gemm_wmma_kernel<0><<<blocks_for(256),  256, 0, stream>>>(aoutH, WoutP, bout, src2b, kL, 256, 256);
    ln_kernel<1><<<(kL + 7) / 8, 256, 0, stream>>>(src, src2b, g1, be1, xb, xf16, kL);

    // 7) FFN: ReLU(x @ W1 + b1) in f16, then @ W2 + b2 in f32
    gemm_wmma_kernel<1><<<blocks_for(1024), 256, 0, stream>>>(xf16, W1P, b1, hb,   kL, kDFF, 256);
    gemm_wmma_kernel<0><<<blocks_for(256),  256, 0, stream>>>(hb,   W2P, b2, ffnb, kL, 256, kDFF);

    // 8) residual + LN2 -> final output
    ln_kernel<0><<<(kL + 7) / 8, 256, 0, stream>>>(xb, ffnb, g2, be2, out, nullptr, kL);
}
